// Attention_42202348650768
// MI455X (gfx1250) — matrix operand order repaired and checked by reading, not compiled, not run on hardware
//
#include <hip/hip_runtime.h>
#include <hip/hip_bf16.h>

// ---- CDNA5 (gfx1250) vector types for WMMA operands -------------------------
typedef __attribute__((ext_vector_type(16))) __bf16   v16bf;
typedef __attribute__((ext_vector_type(8)))  float    v8f;
typedef __attribute__((ext_vector_type(4)))  float    v4f;
typedef __attribute__((ext_vector_type(8)))  unsigned v8u;

using u32 = unsigned int;
using u16 = unsigned short;
using u64 = unsigned long long;

// Problem constants (match reference: B=16, S=1024, D=768, H=12, Dh=64)
constexpr int Bb = 16;
constexpr int S  = 1024;
constexpr int D  = 768;
constexpr int H  = 12;
constexpr int Dh = 64;

// round-to-nearest-even f32 -> bf16
static __device__ __forceinline__ u16 f2bf(float f) {
  u32 u = __builtin_bit_cast(u32, f);
  u32 r = u + 0x7FFFu + ((u >> 16) & 1u);
  return (u16)(r >> 16);
}
static __device__ __forceinline__ u32 pack2bf(float lo, float hi) {
  return (u32)f2bf(lo) | ((u32)f2bf(hi) << 16);
}

// CDNA5 wave32 WMMA register layouts (bf16 16x16x32):
//   A (16x32): lane m=L%16, g=L/16; VGPR j holds K = 16*(j/4)+2*(j%4)+8*g (+1)
//   B (32x16): lane n=L%16, g=L/16; this code loads A and B alike: VGPR j holds K = 16*g+2*j (+1)
//   C/D f32  : VGPR r -> row M = r+8*g, col N = L%16

// Async copy global -> LDS (CDNA5 ASYNCcnt path). lds_off is the raw LDS byte
// offset (dynamic LDS starts at offset 0; no static LDS in the kernels using it).
static __device__ __forceinline__ void async_g2l_b128(u32 lds_off, u64 gaddr) {
  asm volatile("global_load_async_to_lds_b128 %0, %1, off"
               :: "v"(lds_off), "v"(gaddr) : "memory");
}
static __device__ __forceinline__ void wait_asynccnt0() {
  asm volatile("s_wait_asynccnt 0x0" ::: "memory");
}

// ---- Kernel 0: f32 -> bf16 conversion ---------------------------------------
__global__ void cvt_f32_bf16(const float* __restrict__ in, u16* __restrict__ out, int n) {
  int i = blockIdx.x * blockDim.x + threadIdx.x;
  if (i < n) out[i] = f2bf(in[i]);
}

// ---- Kernel 1: QKV projection GEMM ------------------------------------------
// Each wave computes 32 rows x 64 cols (2 M-tiles reuse every B load); the 4
// waves of a block share the same 64-column W strip (L1 reuse).
__global__ __launch_bounds__(128) void qkv_gemm(const u32* __restrict__ xb,
                                                const u32* __restrict__ wb,
                                                u16* __restrict__ q,
                                                u16* __restrict__ k,
                                                u16* __restrict__ vt) {
  const int widx = threadIdx.x >> 5;
  const int lane = threadIdx.x & 31;
  const int n4   = blockIdx.x % (3 * D / 64);                // 36 col-strips
  const int mt2  = (blockIdx.x / (3 * D / 64)) * 4 + widx;   // 512 row-strips of 32
  const int m    = lane & 15;
  const int g    = lane >> 4;

  const u32* arow0 = xb + (size_t)(mt2 * 32 + m) * (D / 2);
  const u32* arow1 = arow0 + 16 * (D / 2);
  v8f acc[2][4] = {};

  for (int kk0 = 0; kk0 < D; kk0 += 32) {
    __builtin_prefetch(arow0 + (kk0 >> 1) + 16, 0, 0);   // global_prefetch_b8
    __builtin_prefetch(arow1 + (kk0 >> 1) + 16, 0, 0);
    v8u ap0, ap1;
#pragma unroll
    for (int j = 0; j < 8; ++j) {
      const int idx = (kk0 >> 1) + 8 * g + j;
      ap0[j] = arow0[idx];
      ap1[j] = arow1[idx];
    }
    v16bf a0 = __builtin_bit_cast(v16bf, ap0);
    v16bf a1 = __builtin_bit_cast(v16bf, ap1);
#pragma unroll
    for (int t = 0; t < 4; ++t) {
      const u32* brow = wb + (size_t)(n4 * 64 + t * 16 + m) * (D / 2);
      v8u bp;
#pragma unroll
      for (int j = 0; j < 8; ++j) bp[j] = brow[(kk0 >> 1) + 8 * g + j];
      v16bf b = __builtin_bit_cast(v16bf, bp);
      acc[0][t] = __builtin_amdgcn_wmma_f32_16x16x32_bf16(false, a0, false, b,
                                                          (short)0, acc[0][t], false, false);
      acc[1][t] = __builtin_amdgcn_wmma_f32_16x16x32_bf16(false, a1, false, b,
                                                          (short)0, acc[1][t], false, false);
    }
  }

#pragma unroll
  for (int mi = 0; mi < 2; ++mi) {
#pragma unroll
    for (int t = 0; t < 4; ++t) {
#pragma unroll
      for (int r = 0; r < 8; ++r) {
        const int mrow = mt2 * 32 + mi * 16 + r + 8 * g;
        const int e    = n4 * 64 + t * 16 + m;     // column in [0, 2304)
        const int which = e / D;                   // uniform within the 16-span
        const int d  = e % D;
        const int h  = d >> 6;
        const int dh = d & 63;
        const int bi = mrow >> 10;
        const int s  = mrow & 1023;
        const float v = acc[mi][t][r];
        if (which == 0)
          q[(((size_t)(bi * H + h)) * S + s) * Dh + dh] = f2bf(v * 0.125f); // 1/sqrt(64)
        else if (which == 1)
          k[(((size_t)(bi * H + h)) * S + s) * Dh + dh] = f2bf(v);
        else
          vt[(((size_t)(bi * H + h)) * Dh + dh) * S + s] = f2bf(v);
      }
    }
  }
}

// ---- Kernel 2: attention core (scores + softmax + PV) -----------------------
// 4 waves/WG, 64 q rows/WG. K and V tiles staged once per 64-k chunk in LDS via
// async global->LDS copies, DOUBLE-BUFFERED so tile c+1 streams in while all
// waves compute on tile c. V chunk 0 is prefetched before the softmax so the
// async engine overlaps the softmax scans.
// Dynamic LDS: 4 x 64KB f32 score strips + 2 x 8KB K/V tiles = 278,528 B.
constexpr int ATTN_STRIP_BYTES = 16 * S * 4;                // 64 KB per wave
constexpr int ATTN_KV_OFF      = 4 * ATTN_STRIP_BYTES;      // 262144
constexpr int ATTN_TILE_BYTES  = 64 * Dh * 2;               // 8 KB
constexpr int ATTN_LDS_BYTES   = ATTN_KV_OFF + 2 * ATTN_TILE_BYTES;

__global__ __launch_bounds__(128) void attn_core(const u16* __restrict__ q,
                                                 const u16* __restrict__ k,
                                                 const u16* __restrict__ vt,
                                                 u16* __restrict__ o) {
  extern __shared__ char smem[];

  const int tid  = threadIdx.x;
  const int w    = tid >> 5;
  const int lane = tid & 31;
  const int m    = lane & 15;
  const int g    = lane >> 4;
  const int qblk = blockIdx.x & (S / 64 - 1);   // 16 blocks of 64 q rows per head
  const int bh   = blockIdx.x >> 4;             // b*H + h
  const int q0   = qblk * 64 + w * 16;          // this wave's 16 q rows

  float* sc = (float*)smem + w * 16 * S;        // this wave's score strip

  // Preload Q A-operands (Dh=64 -> two 16x32 chunks); q pre-scaled by 1/8.
  const u32* qrow = (const u32*)(q + ((size_t)bh * S + q0 + m) * Dh);
  v16bf aQ[2];
#pragma unroll
  for (int c = 0; c < 2; ++c) {
    v8u ap;
#pragma unroll
    for (int j = 0; j < 8; ++j)
      ap[j] = qrow[c * 16 + 8 * g + j];
    aQ[c] = __builtin_bit_cast(v16bf, ap);
  }

  const u64 kglob  = (u64)(const char*)(k  + (size_t)bh * S * Dh);
  const u64 vtglob = (u64)(const char*)(vt + (size_t)bh * Dh * S);

  // K chunk [kt..kt+64) x Dh is one contiguous 8KB block.
  auto issueK = [&](int kt, int buf) {
#pragma unroll
    for (int it = 0; it < 4; ++it) {
      const u32 lin = (u32)(it * 128 + tid) * 16;   // 128 lanes x 16B x 4 = 8KB
      async_g2l_b128((u32)(ATTN_KV_OFF + buf * ATTN_TILE_BYTES) + lin,
                     kglob + (u64)kt * (Dh * 2) + lin);
    }
  };
  // Vt tile: rows dh (stride S*2 bytes), 128B row segment per chunk.
  auto issueV = [&](int kt, int buf) {
#pragma unroll
    for (int it = 0; it < 4; ++it) {
      const u32 lin = (u32)(it * 128 + tid);        // 512 x 16B segments
      const u32 dh  = lin >> 3;
      const u32 seg = lin & 7;
      async_g2l_b128((u32)(ATTN_KV_OFF + buf * ATTN_TILE_BYTES) + dh * (64 * 2) + seg * 16,
                     vtglob + (u64)dh * (S * 2) + (u64)kt * 2 + seg * 16);
    }
  };

  // ---- Phase 1: scores, double-buffered K tiles ----
  issueK(0, 0);
  for (int c = 0; c < S / 64; ++c) {
    wait_asynccnt0();
    __syncthreads();                              // tile c visible to all waves
    if (c + 1 < S / 64) issueK((c + 1) * 64, (c + 1) & 1);   // stream next tile
    const u32* kb = (const u32*)(smem + ATTN_KV_OFF + (c & 1) * ATTN_TILE_BYTES);
    const int  kt = c * 64;
#pragma unroll
    for (int sub = 0; sub < 4; ++sub) {           // 4 x 16 k-columns
      v8f cacc = {};
#pragma unroll
      for (int cc = 0; cc < 2; ++cc) {            // Dh split into 2 x 32
        v8u bp;
#pragma unroll
        for (int j = 0; j < 8; ++j)
          bp[j] = kb[(sub * 16 + m) * (Dh / 2) + cc * 16 + 8 * g + j];
        v16bf b = __builtin_bit_cast(v16bf, bp);
        cacc = __builtin_amdgcn_wmma_f32_16x16x32_bf16(false, aQ[cc], false, b,
                                                       (short)0, cacc, false, false);
      }
#pragma unroll
      for (int r = 0; r < 8; ++r)
        sc[(r + 8 * g) * S + kt + sub * 16 + m] = cacc[r];
    }
  }

  // Prefetch V chunk 0 now so it streams during the softmax.
  issueV(0, 0);

  // ---- Phase 2: softmax. 2 lanes/row (halves), float4 scans, wave32 shuffle.
  // Normalization (1/sum) is NOT applied here; it is folded into the bf16 pack
  // of the P operand below (lane->row mapping matches the A layout).
  float inv;
  {
    v4f* rowv = (v4f*)(sc + m * S + g * (S / 2));
    float mx = -3.0e38f;
    for (int i = 0; i < S / 8; ++i) {             // 128 x float4
      v4f v = rowv[i];
      mx = fmaxf(mx, fmaxf(fmaxf(v.x, v.y), fmaxf(v.z, v.w)));
    }
    mx = fmaxf(mx, __shfl_xor(mx, 16, 32));
    float sum = 0.f;
    for (int i = 0; i < S / 8; ++i) {
      v4f v = rowv[i];
      v.x = __expf(v.x - mx);
      v.y = __expf(v.y - mx);
      v.z = __expf(v.z - mx);
      v.w = __expf(v.w - mx);
      sum += (v.x + v.y) + (v.z + v.w);
      rowv[i] = v;
    }
    sum += __shfl_xor(sum, 16, 32);
    inv = 1.0f / sum;
  }

  // ---- Phase 3: O = P @ V, double-buffered Vt tiles [64 dh][64 s] ----
  v8f acc[4] = {};
  for (int c = 0; c < S / 64; ++c) {
    wait_asynccnt0();
    __syncthreads();
    if (c + 1 < S / 64) issueV((c + 1) * 64, (c + 1) & 1);
    const u32* vb = (const u32*)(smem + ATTN_KV_OFF + (c & 1) * ATTN_TILE_BYTES);
    const int  kt = c * 64;
#pragma unroll
    for (int ks = 0; ks < 64; ks += 32) {
      v8u ap;
#pragma unroll
      for (int j = 0; j < 8; ++j) {
        const int kb = 16 * g + 2 * j;
        ap[j] = pack2bf(sc[m * S + kt + ks + kb] * inv,
                        sc[m * S + kt + ks + kb + 1] * inv);
      }
      v16bf a = __builtin_bit_cast(v16bf, ap);
#pragma unroll
      for (int t = 0; t < 4; ++t) {
        v8u bp;
#pragma unroll
        for (int j = 0; j < 8; ++j)
          bp[j] = vb[(t * 16 + m) * 32 + (ks >> 1) + 8 * g + j];
        v16bf b = __builtin_bit_cast(v16bf, bp);
        acc[t] = __builtin_amdgcn_wmma_f32_16x16x32_bf16(false, a, false, b,
                                                         (short)0, acc[t], false, false);
      }
    }
  }

  // ---- store O as bf16 [B, S, D] with head offset ----
  const int bi = bh / H;
  const int h  = bh % H;
#pragma unroll
  for (int t = 0; t < 4; ++t) {
#pragma unroll
    for (int r = 0; r < 8; ++r) {
      const int srow = q0 + r + 8 * g;
      const int dcol = h * 64 + t * 16 + m;
      o[((size_t)bi * S + srow) * D + dcol] = f2bf(acc[t][r]);
    }
  }
}

// ---- Kernel 3: output projection + bias -------------------------------------
// out[16384,768] = O[16384,768] @ W_out[768,768]^T + b_out. Same 32x64-per-wave
// blocking as qkv_gemm.
__global__ __launch_bounds__(128) void out_gemm(const u32* __restrict__ ob,
                                                const u32* __restrict__ wb,
                                                const float* __restrict__ bias,
                                                float* __restrict__ out) {
  const int widx = threadIdx.x >> 5;
  const int lane = threadIdx.x & 31;
  const int n4   = blockIdx.x % (D / 64);                  // 12 col-strips
  const int mt2  = (blockIdx.x / (D / 64)) * 4 + widx;     // 512 row-strips of 32
  const int m    = lane & 15;
  const int g    = lane >> 4;

  const u32* arow0 = ob + (size_t)(mt2 * 32 + m) * (D / 2);
  const u32* arow1 = arow0 + 16 * (D / 2);
  v8f acc[2][4] = {};

  for (int kk0 = 0; kk0 < D; kk0 += 32) {
    __builtin_prefetch(arow0 + (kk0 >> 1) + 16, 0, 0);
    __builtin_prefetch(arow1 + (kk0 >> 1) + 16, 0, 0);
    v8u ap0, ap1;
#pragma unroll
    for (int j = 0; j < 8; ++j) {
      const int idx = (kk0 >> 1) + 8 * g + j;
      ap0[j] = arow0[idx];
      ap1[j] = arow1[idx];
    }
    v16bf a0 = __builtin_bit_cast(v16bf, ap0);
    v16bf a1 = __builtin_bit_cast(v16bf, ap1);
#pragma unroll
    for (int t = 0; t < 4; ++t) {
      const u32* brow = wb + (size_t)(n4 * 64 + t * 16 + m) * (D / 2);
      v8u bp;
#pragma unroll
      for (int j = 0; j < 8; ++j) bp[j] = brow[(kk0 >> 1) + 8 * g + j];
      v16bf b = __builtin_bit_cast(v16bf, bp);
      acc[0][t] = __builtin_amdgcn_wmma_f32_16x16x32_bf16(false, a0, false, b,
                                                          (short)0, acc[0][t], false, false);
      acc[1][t] = __builtin_amdgcn_wmma_f32_16x16x32_bf16(false, a1, false, b,
                                                          (short)0, acc[1][t], false, false);
    }
  }

#pragma unroll
  for (int mi = 0; mi < 2; ++mi) {
#pragma unroll
    for (int t = 0; t < 4; ++t) {
#pragma unroll
      for (int r = 0; r < 8; ++r) {
        const int mrow = mt2 * 32 + mi * 16 + r + 8 * g;
        const int ncol = n4 * 64 + t * 16 + m;
        out[(size_t)mrow * D + ncol] = acc[mi][t][r] + bias[ncol];
      }
    }
  }
}

// ---- launcher ---------------------------------------------------------------
extern "C" void kernel_launch(void* const* d_in, const int* in_sizes, int n_in,
                              void* d_out, int out_size, void* d_ws, size_t ws_size,
                              hipStream_t stream) {
  const float* x     = (const float*)d_in[0];   // [16,1024,768]
  const float* W_qkv = (const float*)d_in[1];   // [2304,768]
  const float* W_out = (const float*)d_in[2];   // [768,768]
  const float* b_out = (const float*)d_in[3];   // [768]

  constexpr size_t X_ELEMS    = (size_t)Bb * S * D;        // 12,582,912
  constexpr size_t WQKV_ELEMS = (size_t)3 * D * D;         //  1,769,472
  constexpr size_t WOUT_ELEMS = (size_t)D * D;             //    589,824
  constexpr size_t HEAD_ELEMS = (size_t)Bb * H * S * Dh;   // 12,582,912

  char* ws = (char*)d_ws;
  size_t off = 0;
  u16* xb    = (u16*)(ws + off); off += X_ELEMS * 2;
  u16* wqkvb = (u16*)(ws + off); off += WQKV_ELEMS * 2;
  u16* woutb = (u16*)(ws + off); off += WOUT_ELEMS * 2;
  u16* qb    = (u16*)(ws + off); off += HEAD_ELEMS * 2;
  u16* kb    = (u16*)(ws + off); off += HEAD_ELEMS * 2;
  u16* vtb   = (u16*)(ws + off); off += HEAD_ELEMS * 2;
  u16* ob    = (u16*)(ws + off); off += X_ELEMS * 2;       // total ~125 MB

  // 0) precision downcast to bf16 (RNE)
  cvt_f32_bf16<<<(int)((X_ELEMS + 255) / 256), 256, 0, stream>>>(x, xb, (int)X_ELEMS);
  cvt_f32_bf16<<<(int)((WQKV_ELEMS + 255) / 256), 256, 0, stream>>>(W_qkv, wqkvb, (int)WQKV_ELEMS);
  cvt_f32_bf16<<<(int)((WOUT_ELEMS + 255) / 256), 256, 0, stream>>>(W_out, woutb, (int)WOUT_ELEMS);

  // 1) qkv projection: 512 row-strips x 36 col-strips, 4 waves/block
  qkv_gemm<<<(512 / 4) * 36, 128, 0, stream>>>((const u32*)xb, (const u32*)wqkvb, qb, kb, vtb);

  // 2) attention: B*H*(S/64) = 3072 four-wave workgroups, 272KB dynamic LDS
  attn_core<<<Bb * H * (S / 64), 128, ATTN_LDS_BYTES, stream>>>(qb, kb, vtb, ob);

  // 3) output projection: 512 row-strips x 12 col-strips, 4 waves/block
  out_gemm<<<(512 / 4) * 12, 128, 0, stream>>>((const u32*)ob, (const u32*)woutb, b_out, (float*)d_out);
}
